// affine_coupling_layer_89833535963578
// MI455X (gfx1250) — matrix operand order repaired and checked by reading, not compiled, not run on hardware
//
#include <hip/hip_runtime.h>
#include <hip/hip_bf16.h>

// ---------------------------------------------------------------------------
// Affine coupling layer for MI455X (gfx1250, wave32, WMMA).
// B=8192, D=2048, HALF=1024, C=64, H=4096, BLOCKS=2, K1=HALF+C=1088.
// GEMMs run in bf16 via v_wmma_f32_16x16x32_bf16 (f32 accumulate).
// Tile fills use GLOBAL_LOAD_ASYNC_TO_LDS_B128 (ASYNCcnt) when available.
// ---------------------------------------------------------------------------

#define B_    8192
#define D_    2048
#define HALF_ 1024
#define C_    64
#define H_    4096
#define K1_   1088   // HALF_ + C_

typedef __attribute__((ext_vector_type(8)))  __bf16 v8bf;
typedef __attribute__((ext_vector_type(16))) __bf16 v16bf;
typedef __attribute__((ext_vector_type(8)))  float  v8f;

#if __has_builtin(__builtin_amdgcn_global_load_async_to_lds_b128)
#define USE_ASYNC_LDS 1
typedef int v4i_vs __attribute__((vector_size(16)));
typedef v4i_vs __attribute__((address_space(1)))* as1_v4i;   // global
typedef v4i_vs __attribute__((address_space(3)))* as3_v4i;   // LDS
__device__ __forceinline__ void wait_async0() {
#if __has_builtin(__builtin_amdgcn_s_wait_asynccnt)
    __builtin_amdgcn_s_wait_asynccnt(0);
#else
    asm volatile("s_wait_asynccnt 0x0" ::: "memory");
#endif
}
__device__ __forceinline__ void async_copy16(const __bf16* g, __bf16* s) {
    __builtin_amdgcn_global_load_async_to_lds_b128((as1_v4i)g, (as3_v4i)s, 0, 0);
}
#else
#define USE_ASYNC_LDS 0
__device__ __forceinline__ void wait_async0() {}
#endif

// ---------------------------------------------------------------------------
// Utility kernels
// ---------------------------------------------------------------------------

__global__ void copy_f32x4_kernel(float4* __restrict__ dst,
                                  const float4* __restrict__ src, size_t n4) {
    size_t i = (size_t)blockIdx.x * blockDim.x + threadIdx.x;
    if (i < n4) dst[i] = src[i];
}

// f32 [K][N] row-major  ->  bf16 [N][K] row-major (transposed, converted)
__global__ void transpose_convert_kernel(const float* __restrict__ W,
                                         __bf16* __restrict__ WT,
                                         int K, int N) {
    __shared__ float tile[32][33];
    int nx = blockIdx.x * 32 + threadIdx.x;   // N index
    int ky = blockIdx.y * 32 + threadIdx.y;   // K index (threadIdx.y = 0..7)
#pragma unroll
    for (int j = 0; j < 32; j += 8)
        tile[threadIdx.y + j][threadIdx.x] = W[(size_t)(ky + j) * N + nx];
    __syncthreads();
    int kx = blockIdx.y * 32 + threadIdx.x;   // K index
    int ny = blockIdx.x * 32 + threadIdx.y;   // N index
#pragma unroll
    for (int j = 0; j < 32; j += 8)
        WT[(size_t)(ny + j) * K + kx] = (__bf16)tile[threadIdx.x][threadIdx.y + j];
}

// A0[b, j] = bf16( j < HALF ? x[b, idx_id[j]] : cond[b, j-HALF] )
__global__ void pack_input_kernel(__bf16* __restrict__ A0,
                                  const float* __restrict__ x,
                                  const float* __restrict__ cond,
                                  const int* __restrict__ idx_id) {
    int b = blockIdx.y;
    int j = blockIdx.x * 256 + threadIdx.x;
    if (j >= K1_) return;
    float v = (j < HALF_) ? x[(size_t)b * D_ + idx_id[j]]
                          : cond[(size_t)b * C_ + (j - HALF_)];
    A0[(size_t)b * K1_ + j] = (__bf16)v;
}

// x[b, idx_tr[j]] = x[b, idx_tr[j]] * exp(S_j) + T_j ; logdet[b] (+)= sum_j S_j
__global__ void update_kernel(float* __restrict__ xout,
                              float* __restrict__ logdet,
                              const float* __restrict__ out3,
                              const int* __restrict__ idx_tr, int blk) {
    int b   = blockIdx.x;
    int tid = threadIdx.x;
    const float* row  = out3 + (size_t)b * D_;
    float*       xrow = xout + (size_t)b * D_;
    float ssum = 0.f;
#pragma unroll
    for (int j = tid; j < HALF_; j += 256) {
        float S = row[2 * j];
        float T = row[2 * j + 1];
        int   c = idx_tr[j];
        float xv = xrow[c];
        xrow[c] = xv * __expf(S) + T;
        ssum += S;
    }
#pragma unroll
    for (int off = 16; off > 0; off >>= 1)
        ssum += __shfl_down(ssum, off, 32);
    __shared__ float ws[8];
    if ((tid & 31) == 0) ws[tid >> 5] = ssum;
    __syncthreads();
    if (tid == 0) {
        float t = 0.f;
#pragma unroll
        for (int w = 0; w < 8; w++) t += ws[w];
        logdet[b] = (blk == 0 ? 0.f : logdet[b]) + t;
    }
}

// ---------------------------------------------------------------------------
// WMMA bf16 GEMM:  C[M,N] = act(A[M,K] * BT[N,K]^T + bias)
//   A  : bf16 row-major [M, K]
//   BT : bf16 row-major [N, K]   (W transposed, K contiguous)
//   out: bf16 (RELU path) or f32
// Workgroup: 256 threads (8 waves). Tile 128x128, BK=64, double-buffered LDS.
// Each wave computes a 32x64 tile = 2x4 grid of 16x16 f32 accumulators.
// Pipeline per K-tile: wait_asynccnt -> barrier -> issue async fill of next
// buffer -> 16 WMMAs on current buffer (copy/compute fully overlapped).
// ---------------------------------------------------------------------------

#define BM   128
#define BN   128
#define BK   64
#define KPAD 72   // BK + 8 bf16 pad (16B) to stagger LDS banks

// Assemble a 16-element bf16 fragment from two 16-byte LDS reads.
__device__ __forceinline__ v16bf make_frag(const __bf16* p0, const __bf16* p1) {
    v8bf lo = *(const v8bf*)p0;
    v8bf hi = *(const v8bf*)p1;
    return __builtin_shufflevector(lo, hi, 0, 1, 2, 3, 4, 5, 6, 7,
                                   8, 9, 10, 11, 12, 13, 14, 15);
}

template <bool RELU, bool OUT_BF16>
__global__ __launch_bounds__(256)
void gemm_bf16_wmma_kernel(const __bf16* __restrict__ A,
                           const __bf16* __restrict__ BT,
                           const float* __restrict__ bias,
                           void* __restrict__ Cout,
                           int M, int N, int K) {
    __shared__ __bf16 As[2][BM][KPAD];
    __shared__ __bf16 Bs[2][BN][KPAD];

    const int tid  = threadIdx.x;
    const int lane = tid & 31;
    const int wave = tid >> 5;
    const int wm   = wave >> 1;   // 0..3  (M direction)
    const int wn   = wave & 1;    // 0..1  (N direction)
    const int bm   = blockIdx.y * BM;
    const int bn   = blockIdx.x * BN;

    // Cooperative tile fill: each thread moves 32 bf16 (64B) per matrix.
    const int lrow = tid >> 1;            // 0..127
    const int lcol = (tid & 1) * 32;      // 0 or 32

    const __bf16* aptr = A  + (size_t)(bm + lrow) * K + lcol;
    const __bf16* bptr = BT + (size_t)(bn + lrow) * K + lcol;

    v8f acc[2][4];
#pragma unroll
    for (int i = 0; i < 2; i++)
#pragma unroll
        for (int j = 0; j < 4; j++) acc[i][j] = (v8f){};

    const int ntiles = K / BK;

    auto issue_tile = [&](int buf, int kt) {
        const __bf16* ga = aptr + kt;
        const __bf16* gb = bptr + kt;
        __bf16* sa = &As[buf][lrow][lcol];
        __bf16* sb = &Bs[buf][lrow][lcol];
#if USE_ASYNC_LDS
#pragma unroll
        for (int i = 0; i < 4; i++) {
            async_copy16(ga + i * 8, sa + i * 8);   // 16B per lane per op
            async_copy16(gb + i * 8, sb + i * 8);
        }
#else
        uint4 ra[4], rb[4];
#pragma unroll
        for (int i = 0; i < 4; i++) {
            ra[i] = ((const uint4*)ga)[i];
            rb[i] = ((const uint4*)gb)[i];
        }
#pragma unroll
        for (int i = 0; i < 4; i++) {
            ((uint4*)sa)[i] = ra[i];
            ((uint4*)sb)[i] = rb[i];
        }
#endif
    };

    auto compute = [&](int buf) {
#pragma unroll
        for (int kk = 0; kk < BK; kk += 32) {
            v16bf af[2], bfr[4];
#pragma unroll
            for (int i = 0; i < 2; i++) {
                // A frag 16x32: lane = (half, m); per lane K runs
                // [kk+half*8, +8) and [kk+16+half*8, +8).
                const __bf16* base =
                    &As[buf][wm * 32 + i * 16 + (lane & 15)][kk + (lane >> 4) * 8];
                af[i] = make_frag(base, base + 16);
            }
#pragma unroll
            for (int j = 0; j < 4; j++) {
                // B frag 32x16: lane = (half, n); per lane K runs
                // [kk+half*8, +8) and [kk+16+half*8, +8).
                const __bf16* base =
                    &Bs[buf][wn * 64 + j * 16 + (lane & 15)][kk + (lane >> 4) * 8];
                bfr[j] = make_frag(base, base + 16);
            }
#pragma unroll
            for (int i = 0; i < 2; i++)
#pragma unroll
                for (int j = 0; j < 4; j++)
                    acc[i][j] = __builtin_amdgcn_wmma_f32_16x16x32_bf16(
                        false, af[i], false, bfr[j], (short)0, acc[i][j],
                        false, false);
        }
    };

    issue_tile(0, 0);
    for (int t = 0; t < ntiles; t++) {
        int cur = t & 1;
        wait_async0();        // this wave's fill of buf `cur` complete
        __syncthreads();      // all waves' fills done; prev compute done (WAR)
        if (t + 1 < ntiles) issue_tile(cur ^ 1, (t + 1) * BK);
        compute(cur);
    }

    // Epilogue: C/D layout -> lane n = lane&15, m = r + (lane>>4)*8 per VGPR r.
#pragma unroll
    for (int i = 0; i < 2; i++) {
#pragma unroll
        for (int j = 0; j < 4; j++) {
            int n     = bn + wn * 64 + j * 16 + (lane & 15);
            int mbase = bm + wm * 32 + i * 16 + (lane >> 4) * 8;
            float bv = bias[n];
#pragma unroll
            for (int r = 0; r < 8; r++) {
                float v = acc[i][j][r] + bv;
                if (RELU) v = fmaxf(v, 0.f);
                size_t off = (size_t)(mbase + r) * N + n;
                if (OUT_BF16) ((__bf16*)Cout)[off] = (__bf16)v;
                else          ((float*)Cout)[off]  = v;
            }
        }
    }
}

static void run_gemm(const __bf16* A, const __bf16* BT, const float* bias,
                     void* out, int M, int N, int K, bool relu_bf16,
                     hipStream_t s) {
    dim3 grid(N / BN, M / BM), blk(256);
    if (relu_bf16)
        gemm_bf16_wmma_kernel<true, true><<<grid, blk, 0, s>>>(A, BT, bias, out, M, N, K);
    else
        gemm_bf16_wmma_kernel<false, false><<<grid, blk, 0, s>>>(A, BT, bias, out, M, N, K);
}

// ---------------------------------------------------------------------------
// Launch
// ---------------------------------------------------------------------------

extern "C" void kernel_launch(void* const* d_in, const int* in_sizes, int n_in,
                              void* d_out, int out_size, void* d_ws, size_t ws_size,
                              hipStream_t stream) {
    (void)in_sizes; (void)n_in; (void)out_size; (void)ws_size;

    const float* z      = (const float*)d_in[0];
    const float* cond   = (const float*)d_in[1];
    const int*   idx_id = (const int*)d_in[2];
    const int*   idx_tr = (const int*)d_in[3];
    const float* W1     = (const float*)d_in[4];
    const float* b1     = (const float*)d_in[5];
    const float* W2     = (const float*)d_in[6];
    const float* b2     = (const float*)d_in[7];
    const float* W3     = (const float*)d_in[8];
    const float* b3     = (const float*)d_in[9];

    float* xout   = (float*)d_out;                 // [B, D]
    float* logdet = xout + (size_t)B_ * D_;        // [B]

    char* ws = (char*)d_ws;
    size_t off = 0;
    auto alloc = [&](size_t bytes) -> void* {
        void* p = ws + off;
        off = (off + bytes + 255) & ~(size_t)255;
        return p;
    };
    __bf16* A0   = (__bf16*)alloc((size_t)B_ * K1_ * 2);
    __bf16* H1   = (__bf16*)alloc((size_t)B_ * H_ * 2);
    __bf16* H2   = (__bf16*)alloc((size_t)B_ * H_ * 2);
    float*  OUT3 = (float*)alloc((size_t)B_ * D_ * 4);
    __bf16* W1T  = (__bf16*)alloc((size_t)H_ * K1_ * 2);
    __bf16* W2T  = (__bf16*)alloc((size_t)H_ * H_ * 2);
    __bf16* W3T  = (__bf16*)alloc((size_t)D_ * H_ * 2);

    // x <- z
    {
        size_t n4 = (size_t)B_ * D_ / 4;
        copy_f32x4_kernel<<<dim3((unsigned)((n4 + 255) / 256)), dim3(256), 0, stream>>>(
            (float4*)xout, (const float4*)z, n4);
    }

    dim3 tb(32, 8);
    for (int blk = 0; blk < 2; blk++) {
        // Weight transpose + f32->bf16 (weights stay L2-resident afterwards).
        transpose_convert_kernel<<<dim3(H_ / 32, K1_ / 32), tb, 0, stream>>>(
            W1 + (size_t)blk * K1_ * H_, W1T, K1_, H_);
        transpose_convert_kernel<<<dim3(H_ / 32, H_ / 32), tb, 0, stream>>>(
            W2 + (size_t)blk * H_ * H_, W2T, H_, H_);
        transpose_convert_kernel<<<dim3(D_ / 32, H_ / 32), tb, 0, stream>>>(
            W3 + (size_t)blk * H_ * D_, W3T, H_, D_);

        // h0 = bf16(concat(x[:, idx_id], cond))
        pack_input_kernel<<<dim3((K1_ + 255) / 256, B_), dim3(256), 0, stream>>>(
            A0, xout, cond, idx_id + blk * HALF_);

        // h1 = relu(h0 @ W1 + b1)  [8192 x 4096]
        run_gemm(A0, W1T, b1 + (size_t)blk * H_, H1, B_, H_, K1_, true, stream);
        // h2 = relu(h1 @ W2 + b2)  [8192 x 4096]
        run_gemm(H1, W2T, b2 + (size_t)blk * H_, H2, B_, H_, H_, true, stream);
        // out = h2 @ W3 + b3       [8192 x 2048] f32
        run_gemm(H2, W3T, b3 + (size_t)blk * D_, OUT3, B_, D_, H_, false, stream);

        // x[:, idx_tr] = x[:, idx_tr]*exp(S) + T ; logdet += sum(S)
        update_kernel<<<dim3(B_), dim3(256), 0, stream>>>(
            xout, logdet, OUT3, idx_tr + blk * HALF_, blk);
    }
}
